// Flow_Attention_3427383902646
// MI455X (gfx1250) — matrix operand order repaired and checked by reading, not compiled, not run on hardware
//
#include <hip/hip_runtime.h>
#include <hip/hip_bf16.h>
#include <math.h>

typedef __bf16 bf16;
typedef __bf16 v8bf  __attribute__((ext_vector_type(8)));
typedef __bf16 v16bf __attribute__((ext_vector_type(16)));
typedef float  v8f   __attribute__((ext_vector_type(8)));

static constexpr int Bb = 4, Ll = 4096, Dd = 1024, Hh = 16, DH = 64;
static constexpr float FEPS = 1e-6f;

__device__ __forceinline__ float fast_rcp(float x)     { return __builtin_amdgcn_rcpf(x); }
__device__ __forceinline__ float fast_sigmoid(float x) { return __builtin_amdgcn_rcpf(1.0f + __expf(-x)); }

__device__ __forceinline__ float wave_sum(float v) {
    #pragma unroll
    for (int o = 16; o > 0; o >>= 1) v += __shfl_xor(v, o, 32);
    return v;
}

// ---------------------------------------------------------------------------
// WMMA GEMM: C[M,N] = act(A[M,K] * W[K,N] + bias), fp32 in/out, bf16 compute.
// Block tile 64x128, K-tile 64, double-buffered LDS, 8 waves (2x4),
// each wave 32x32 out (2x2 wmma tiles), 8 v_wmma per K-tile iteration.
// ---------------------------------------------------------------------------
#define SB 72   // LDS row stride in bf16 (64 + 8 pad); keeps 16B alignment

__device__ __forceinline__ void gemm_loadG(
    const float* __restrict__ A, const float* __restrict__ W,
    int K, int N, int by, int bx, int tid, int k0,
    float4 ra[4], float4 rb[8])
{
    #pragma unroll
    for (int i = 0; i < 4; ++i) {               // A tile 64x64: 1024 float4
        int c = tid + i * 256;
        int r = c >> 4, c4 = c & 15;
        ra[i] = *reinterpret_cast<const float4*>(A + (size_t)(by * 64 + r) * K + k0 + c4 * 4);
    }
    #pragma unroll
    for (int i = 0; i < 8; ++i) {               // B tile 64x128: 2048 float4
        int c = tid + i * 256;
        int r = c >> 5, c4 = c & 31;
        rb[i] = *reinterpret_cast<const float4*>(W + (size_t)(k0 + r) * N + bx * 128 + c4 * 4);
    }
}

__device__ __forceinline__ void gemm_storeL(
    bf16* __restrict__ As, bf16* __restrict__ Bs, int tid,
    const float4 ra[4], const float4 rb[8])
{
    #pragma unroll
    for (int i = 0; i < 4; ++i) {               // As[row][k]
        int c = tid + i * 256;
        int r = c >> 4, c4 = c & 15;
        bf16* dst = &As[r * SB + c4 * 4];
        dst[0] = (bf16)ra[i].x; dst[1] = (bf16)ra[i].y;
        dst[2] = (bf16)ra[i].z; dst[3] = (bf16)ra[i].w;
    }
    #pragma unroll
    for (int i = 0; i < 8; ++i) {               // Bs transposed [col][k]
        int c = tid + i * 256;
        int r = c >> 5, c4 = c & 31;
        Bs[(c4 * 4 + 0) * SB + r] = (bf16)rb[i].x;
        Bs[(c4 * 4 + 1) * SB + r] = (bf16)rb[i].y;
        Bs[(c4 * 4 + 2) * SB + r] = (bf16)rb[i].z;
        Bs[(c4 * 4 + 3) * SB + r] = (bf16)rb[i].w;
    }
}

template <int ACT>
__global__ __launch_bounds__(256) void gemm_bf16wmma(
    const float* __restrict__ A, const float* __restrict__ W,
    const float* __restrict__ bias, float* __restrict__ C,
    int M, int K, int N)
{
    __shared__ bf16 As[2][64 * SB];
    __shared__ bf16 Bs[2][128 * SB];
    const int tid  = threadIdx.x;
    const int lane = tid & 31;
    const int wave = tid >> 5;
    const int wm   = wave >> 2;      // 0..1
    const int wn   = wave & 3;       // 0..3
    const int bx   = blockIdx.x;     // N tile (128)
    const int by   = blockIdx.y;     // M tile (64)

    v8f acc[2][2] = {};
    float4 ra[4], rb[8];

    const int nK = K >> 6;
    gemm_loadG(A, W, K, N, by, bx, tid, 0, ra, rb);
    gemm_storeL(As[0], Bs[0], tid, ra, rb);
    __syncthreads();

    for (int kt = 0; kt < nK; ++kt) {
        const int  cur  = kt & 1;
        const bool more = (kt + 1) < nK;
        if (more) gemm_loadG(A, W, K, N, by, bx, tid, (kt + 1) << 6, ra, rb);

        const bf16* Ac = As[cur];
        const bf16* Bc = Bs[cur];
        // CDNA5 16-bit fragment layouts:
        // A 16x32: lanes 0-15 M=lane, K = klo+{0..7} and klo+16+{0..7}, klo = 8*(lane>>4)
        // B 32x16: N=lane&15, K = kb+{0..7} and kb+16+{0..7}, kb = 8*(lane>>4)
        #pragma unroll
        for (int ks = 0; ks < 2; ++ks) {
            v16bf afrag[2], bfrag[2];
            const int klo = ks * 32 + (lane >> 4) * 8;
            #pragma unroll
            for (int mi = 0; mi < 2; ++mi) {
                int row = wm * 32 + mi * 16 + (lane & 15);
                v8bf lo = *reinterpret_cast<const v8bf*>(&Ac[row * SB + klo]);
                v8bf hi = *reinterpret_cast<const v8bf*>(&Ac[row * SB + klo + 16]);
                afrag[mi] = __builtin_shufflevector(lo, hi, 0,1,2,3,4,5,6,7,8,9,10,11,12,13,14,15);
            }
            const int kb = ks * 32 + (lane >> 4) * 8;
            #pragma unroll
            for (int ni = 0; ni < 2; ++ni) {
                int col = wn * 32 + ni * 16 + (lane & 15);
                v8bf lo = *reinterpret_cast<const v8bf*>(&Bc[col * SB + kb]);
                v8bf hi = *reinterpret_cast<const v8bf*>(&Bc[col * SB + kb + 16]);
                bfrag[ni] = __builtin_shufflevector(lo, hi, 0,1,2,3,4,5,6,7,8,9,10,11,12,13,14,15);
            }
            #pragma unroll
            for (int mi = 0; mi < 2; ++mi)
                #pragma unroll
                for (int ni = 0; ni < 2; ++ni)
                    acc[mi][ni] = __builtin_amdgcn_wmma_f32_16x16x32_bf16(
                        false, afrag[mi], false, bfrag[ni], (short)0, acc[mi][ni], false, false);
        }

        if (more) gemm_storeL(As[cur ^ 1], Bs[cur ^ 1], tid, ra, rb);
        __syncthreads();
    }

    // epilogue: bias (+ compile-time sigmoid). C/D layout: VGPR r -> M=r+8*(lane>>4), N=lane&15
    #pragma unroll
    for (int mi = 0; mi < 2; ++mi) {
        #pragma unroll
        for (int ni = 0; ni < 2; ++ni) {
            int colg = bx * 128 + wn * 32 + ni * 16 + (lane & 15);
            float bv = bias[colg];
            #pragma unroll
            for (int r = 0; r < 8; ++r) {
                int rowg = by * 64 + wm * 32 + mi * 16 + r + ((lane >> 4) << 3);
                float v = acc[mi][ni][r] + bv;
                if (ACT) v = fast_sigmoid(v);
                C[(size_t)rowg * N + colg] = v;
            }
        }
    }
}

// ---------------------------------------------------------------------------
__global__ __launch_bounds__(256) void zero_kernel(float* __restrict__ p, int n) {
    int i = blockIdx.x * 256 + threadIdx.x;
    if (i < n) p[i] = 0.0f;
}

// column sums over L: qsum[b, hd] = sum_l qs[b,l,hd]  (and same for k)
__global__ __launch_bounds__(256) void col_sums(
    const float* __restrict__ qs, const float* __restrict__ ks,
    float* __restrict__ qsum, float* __restrict__ ksum)
{
    int b = blockIdx.y, chunk = blockIdx.x, tid = threadIdx.x;
    float aq[4] = {0,0,0,0}, ak[4] = {0,0,0,0};
    for (int r = 0; r < 256; ++r) {
        size_t base = ((size_t)b * Ll + chunk * 256 + r) * Dd;
        #pragma unroll
        for (int j = 0; j < 4; ++j) {
            int col = tid + j * 256;
            aq[j] += qs[base + col];
            ak[j] += ks[base + col];
        }
    }
    #pragma unroll
    for (int j = 0; j < 4; ++j) {
        int col = tid + j * 256;
        atomicAdd(&qsum[b * Dd + col], aq[j]);
        atomicAdd(&ksum[b * Dd + col], ak[j]);
    }
}

// weighted column sums: qsum2 = sum_l qs*sink_in, ksum2 = sum_l ks*src_out
__global__ __launch_bounds__(256) void col_sums2(
    const float* __restrict__ qs, const float* __restrict__ ks,
    const float* __restrict__ sink_in, const float* __restrict__ src_out,
    float* __restrict__ qsum2, float* __restrict__ ksum2)
{
    int b = blockIdx.y, chunk = blockIdx.x, tid = threadIdx.x;
    float aq[4] = {0,0,0,0}, ak[4] = {0,0,0,0};
    for (int r = 0; r < 256; ++r) {
        int l = chunk * 256 + r;
        size_t base = ((size_t)b * Ll + l) * Dd;
        #pragma unroll
        for (int j = 0; j < 4; ++j) {
            int col = tid + j * 256;
            int h = col >> 6;
            size_t o = ((size_t)(b * Hh + h)) * Ll + l;
            aq[j] += qs[base + col] * sink_in[o];
            ak[j] += ks[base + col] * src_out[o];
        }
    }
    #pragma unroll
    for (int j = 0; j < 4; ++j) {
        int col = tid + j * 256;
        atomicAdd(&qsum2[b * Dd + col], aq[j]);
        atomicAdd(&ksum2[b * Dd + col], ak[j]);
    }
}

// one wave per (b,h,l): sink_incoming = 1/dot(q+eps, ksum+eps), source_outgoing likewise
__global__ __launch_bounds__(256) void flow1(
    const float* __restrict__ qs, const float* __restrict__ ks,
    const float* __restrict__ qsum, const float* __restrict__ ksum,
    float* __restrict__ sink_in, float* __restrict__ src_out)
{
    int tid = threadIdx.x, lane = tid & 31, wave = tid >> 5;
    int w  = blockIdx.x * 8 + wave;
    int l  = w & (Ll - 1);
    int bh = w >> 12;                 // Ll = 2^12
    int h  = bh & (Hh - 1), b = bh >> 4;
    size_t rb = ((size_t)b * Ll + l) * Dd + h * DH;
    float q0 = qs[rb + lane], q1 = qs[rb + lane + 32];
    float k0 = ks[rb + lane], k1 = ks[rb + lane + 32];
    const float* KS = ksum + b * Dd + h * DH;
    const float* QS = qsum + b * Dd + h * DH;
    float d1 = (q0 + FEPS) * (KS[lane] + FEPS) + (q1 + FEPS) * (KS[lane + 32] + FEPS);
    float d2 = (k0 + FEPS) * (QS[lane] + FEPS) + (k1 + FEPS) * (QS[lane + 32] + FEPS);
    d1 = wave_sum(d1);
    d2 = wave_sum(d2);
    if (lane == 0) {
        size_t o = (size_t)bh * Ll + l;
        sink_in[o] = fast_rcp(d1);
        src_out[o] = fast_rcp(d2);
    }
}

// conserved_sink -> sink_allocation = sigmoid(cs * L/S); conserved_source -> clip [-1,1]
__global__ __launch_bounds__(256) void flow2(
    const float* __restrict__ qs, const float* __restrict__ ks,
    const float* __restrict__ qsum2, const float* __restrict__ ksum2,
    float* __restrict__ sink_alloc, float* __restrict__ csrc_out)
{
    int tid = threadIdx.x, lane = tid & 31, wave = tid >> 5;
    int w  = blockIdx.x * 8 + wave;
    int l  = w & (Ll - 1);
    int bh = w >> 12;
    int h  = bh & (Hh - 1), b = bh >> 4;
    size_t rb = ((size_t)b * Ll + l) * Dd + h * DH;
    float q0 = qs[rb + lane], q1 = qs[rb + lane + 32];
    float k0 = ks[rb + lane], k1 = ks[rb + lane + 32];
    const float* K2 = ksum2 + b * Dd + h * DH;
    const float* Q2 = qsum2 + b * Dd + h * DH;
    float cs = (q0 + FEPS) * (K2[lane] + FEPS) + (q1 + FEPS) * (K2[lane + 32] + FEPS);
    float cr = (k0 + FEPS) * (Q2[lane] + FEPS) + (k1 + FEPS) * (Q2[lane + 32] + FEPS);
    cs = wave_sum(cs);
    cr = wave_sum(cr);
    if (lane == 0) {
        size_t o = (size_t)bh * Ll + l;
        const float ratio = (float)Ll / (float)Ll;  // L/S
        sink_alloc[o] = fast_sigmoid(cs * ratio);
        csrc_out[o]   = fminf(1.0f, fmaxf(-1.0f, cr));
    }
}

// in-place per-(b,h): p = softmax(p over L) * S
__global__ __launch_bounds__(256) void softmax_scale(float* __restrict__ sc) {
    int bh = blockIdx.x, tid = threadIdx.x;
    __shared__ float red[256];
    float* p = sc + (size_t)bh * Ll;
    float m = -1e30f;
    for (int i = tid; i < Ll; i += 256) m = fmaxf(m, p[i]);
    red[tid] = m; __syncthreads();
    for (int s = 128; s > 0; s >>= 1) { if (tid < s) red[tid] = fmaxf(red[tid], red[tid + s]); __syncthreads(); }
    m = red[0]; __syncthreads();
    float sum = 0.0f;
    for (int i = tid; i < Ll; i += 256) sum += __expf(p[i] - m);
    red[tid] = sum; __syncthreads();
    for (int s = 128; s > 0; s >>= 1) { if (tid < s) red[tid] += red[tid + s]; __syncthreads(); }
    float inv = (float)Ll * fast_rcp(red[0]);
    __syncthreads();
    for (int i = tid; i < Ll; i += 256) p[i] = __expf(p[i] - m) * inv;
}

// per (b,h): kv[d,m] = sum_l ks[l,d] * (v[l,m] * scomp[l])
__global__ __launch_bounds__(256) void kv_kernel(
    const float* __restrict__ ks, const float* __restrict__ v,
    const float* __restrict__ scomp, float* __restrict__ kvbuf)
{
    int bh = blockIdx.x, b = bh >> 4, h = bh & 15;
    int tid = threadIdx.x;
    __shared__ float ksh[8][64];
    __shared__ float vnh[8][64];
    int d = tid & 63, m0 = (tid >> 6) * 16;
    float acc[16];
    #pragma unroll
    for (int j = 0; j < 16; ++j) acc[j] = 0.0f;
    for (int lc = 0; lc < Ll / 8; ++lc) {
        #pragma unroll
        for (int i = 0; i < 2; ++i) {
            int idx = tid + i * 256;
            int rr = idx >> 6, cc = idx & 63;
            int l = lc * 8 + rr;
            size_t base = ((size_t)b * Ll + l) * Dd + h * DH + cc;
            ksh[rr][cc] = ks[base];
            vnh[rr][cc] = v[base] * scomp[(size_t)bh * Ll + l];
        }
        __syncthreads();
        #pragma unroll
        for (int rr = 0; rr < 8; ++rr) {
            float kd = ksh[rr][d];
            #pragma unroll
            for (int j = 0; j < 16; ++j) acc[j] = fmaf(kd, vnh[rr][m0 + j], acc[j]);
        }
        __syncthreads();
    }
    #pragma unroll
    for (int j = 0; j < 16; ++j)
        kvbuf[((size_t)bh * 64 + d) * 64 + m0 + j] = acc[j];
}

// x[l,m] = (sum_d qs[l,d]*sink_in[l] * kv[d,m]) * sink_alloc[l], into xbuf[b,l,h*64+m]
__global__ __launch_bounds__(256) void xout_kernel(
    const float* __restrict__ qs, const float* __restrict__ kvbuf,
    const float* __restrict__ sink_in, const float* __restrict__ sink_alloc,
    float* __restrict__ xbuf)
{
    const int nblk = Ll / 64;
    int bh = blockIdx.x / nblk, chunk = blockIdx.x % nblk;
    int b = bh >> 4, h = bh & 15;
    int tid = threadIdx.x;
    __shared__ float kvs[64][65];
    #pragma unroll
    for (int i = 0; i < 16; ++i) {
        int idx = tid + i * 256;
        kvs[idx >> 6][idx & 63] = kvbuf[(size_t)bh * 4096 + idx];
    }
    __syncthreads();
    int lr = tid >> 2, l = chunk * 64 + lr, m0 = (tid & 3) * 16;
    size_t o = (size_t)bh * Ll + l;
    float si = sink_in[o], sa = sink_alloc[o];
    size_t qb = ((size_t)b * Ll + l) * Dd + h * DH;
    float acc[16];
    #pragma unroll
    for (int j = 0; j < 16; ++j) acc[j] = 0.0f;
    for (int d2 = 0; d2 < 64; ++d2) {
        float qn = qs[qb + d2] * si;
        #pragma unroll
        for (int j = 0; j < 16; ++j) acc[j] = fmaf(qn, kvs[d2][m0 + j], acc[j]);
    }
    #pragma unroll
    for (int j = 0; j < 16; ++j) xbuf[qb + m0 + j] = acc[j] * sa;
}

// ---------------------------------------------------------------------------
extern "C" void kernel_launch(void* const* d_in, const int* in_sizes, int n_in,
                              void* d_out, int out_size, void* d_ws, size_t ws_size,
                              hipStream_t stream)
{
    const float* queries = (const float*)d_in[0];
    const float* keys    = (const float*)d_in[1];
    const float* values  = (const float*)d_in[2];
    const float* Wq = (const float*)d_in[3];
    const float* bq = (const float*)d_in[4];
    const float* Wk = (const float*)d_in[5];
    const float* bk = (const float*)d_in[6];
    const float* Wv = (const float*)d_in[7];
    const float* bv = (const float*)d_in[8];
    const float* Wo = (const float*)d_in[9];
    const float* bo = (const float*)d_in[10];
    float* out = (float*)d_out;

    const size_t MAT = (size_t)Bb * Ll * Dd;       // 16,777,216
    const size_t BHL = (size_t)Bb * Hh * Ll;       // 262,144
    float* ws     = (float*)d_ws;
    float* qproj  = ws;
    float* kproj  = ws + MAT;
    float* vproj  = ws + 2 * MAT;
    float* xbuf   = ws + 3 * MAT;
    float* sums   = ws + 4 * MAT;                  // 4 * 4096 floats
    float* qsum   = sums;
    float* ksum   = sums + Bb * Dd;
    float* qsum2  = sums + 2 * Bb * Dd;
    float* ksum2  = sums + 3 * Bb * Dd;
    float* sink_in = sums + 4 * Bb * Dd;
    float* src_out = sink_in + BHL;
    float* salloc  = src_out + BHL;
    float* scomp   = salloc + BHL;
    float* kvbuf   = scomp + BHL;

    zero_kernel<<<(4 * Bb * Dd + 255) / 256, 256, 0, stream>>>(sums, 4 * Bb * Dd);

    dim3 gg(Dd / 128, (Bb * Ll) / 64);
    gemm_bf16wmma<1><<<gg, 256, 0, stream>>>(queries, Wq, bq, qproj, Bb * Ll, Dd, Dd);
    gemm_bf16wmma<1><<<gg, 256, 0, stream>>>(keys,    Wk, bk, kproj, Bb * Ll, Dd, Dd);
    gemm_bf16wmma<0><<<gg, 256, 0, stream>>>(values,  Wv, bv, vproj, Bb * Ll, Dd, Dd);

    col_sums<<<dim3(Ll / 256, Bb), 256, 0, stream>>>(qproj, kproj, qsum, ksum);
    flow1<<<(Bb * Hh * Ll) / 8, 256, 0, stream>>>(qproj, kproj, qsum, ksum, sink_in, src_out);
    col_sums2<<<dim3(Ll / 256, Bb), 256, 0, stream>>>(qproj, kproj, sink_in, src_out, qsum2, ksum2);
    flow2<<<(Bb * Hh * Ll) / 8, 256, 0, stream>>>(qproj, kproj, qsum2, ksum2, salloc, scomp);
    softmax_scale<<<Bb * Hh, 256, 0, stream>>>(scomp);
    kv_kernel<<<Bb * Hh, 256, 0, stream>>>(kproj, vproj, scomp, kvbuf);
    xout_kernel<<<Bb * Hh * (Ll / 64), 256, 0, stream>>>(qproj, kvbuf, sink_in, salloc, xbuf);

    gemm_bf16wmma<0><<<gg, 256, 0, stream>>>(xbuf, Wo, bo, out, Bb * Ll, Dd, Dd);
}
